// SingleHeadAttention_66254165508299
// MI455X (gfx1250) — matrix operand order repaired and checked by reading, not compiled, not run on hardware
//
#include <hip/hip_runtime.h>
#include <hip/hip_bf16.h>
#include <stdint.h>

typedef __attribute__((ext_vector_type(16))) __bf16 v16bf;
typedef __attribute__((ext_vector_type(8)))  float  v8f;
typedef unsigned short u16;

// ---------- bf16 helpers (round-to-nearest-even) ----------
__device__ __forceinline__ u16 f2bf(float f) {
  uint32_t u = __builtin_bit_cast(uint32_t, f);
  u += 0x7FFFu + ((u >> 16) & 1u);
  return (u16)(u >> 16);
}
__device__ __forceinline__ float bf2f(u16 h) {
  uint32_t u = ((uint32_t)h) << 16;
  return __builtin_bit_cast(float, u);
}

// gfx1250 async global->LDS copy, 16 bytes per lane, tracked by ASYNCcnt.
// VDST = VGPR with LDS byte address (flat->LDS truncates to addr[31:0]),
// VADDR = 64-bit global address, no SADDR.
#define ASYNC_LD_B128(ldsptr, gptr)                                        \
  asm volatile("global_load_async_to_lds_b128 %0, %1, off"                 \
               :: "v"((uint32_t)(uintptr_t)(ldsptr)), "v"(gptr) : "memory")
// Async loads retire in order: ASYNCcnt<=6 completes the stage issued two
// iterations ago while the newest 6 ops remain in flight.
#define WAIT_ASYNC(n) asm volatile("s_wait_asynccnt %0" :: "i"(n) : "memory")

// ---------- f32 -> bf16 bulk convert (vectorized, 4 elems/thread) ----------
__global__ __launch_bounds__(256) void k_cvt(const float* __restrict__ in,
                                             u16* __restrict__ out,
                                             long long n4) {
  long long i = (long long)blockIdx.x * blockDim.x + threadIdx.x;
  if (i >= n4) return;
  float4 f = ((const float4*)in)[i];
  uint2 o;
  o.x = (uint32_t)f2bf(f.x) | ((uint32_t)f2bf(f.y) << 16);
  o.y = (uint32_t)f2bf(f.z) | ((uint32_t)f2bf(f.w) << 16);
  ((uint2*)out)[i] = o;
}

// ---------- NT GEMM: C[M,N] = scale * (A[M,K] @ B[N,K]^T) (+bias, +causal) ----------
// A,B bf16 row-major, K contiguous. Block tile 256x128, K-step 32.
// LDS triple-buffered, filled by GLOBAL_LOAD_ASYNC_TO_LDS_B128 (2-deep pipeline).
// 8 waves in 4(M) x 2(N) grid; wave tile 64x64 = 4x4 WMMA 16x16x32 bf16 tiles.
__global__ __launch_bounds__(256) void k_gemm_nt_bf16(
    const u16* __restrict__ A, const u16* __restrict__ B,
    void* __restrict__ C, const float* __restrict__ bias,
    int M, int N, int K, int lda, int ldb, int ldc,
    long long sA, long long sB, long long sC,
    float scale, int causal, int out_bf16)
{
  __shared__ __align__(16) u16 As[3][256 * 40];  // 256 rows x 32 k, stride 40
  __shared__ __align__(16) u16 Bs[3][128 * 40];  // 128 cols x 32 k, stride 40

  const int tid  = threadIdx.x;
  const int lane = tid & 31;
  const int w    = tid >> 5;
  const int l16  = lane & 15;
  const int hi   = lane >> 4;          // half-wave select
  const int wm   = w >> 1;             // 0..3 -> M offset wm*64
  const int wn   = w & 1;              // 0..1 -> N offset wn*64

  const int m0 = blockIdx.y * 256;
  const int n0 = blockIdx.x * 128;
  const int bz = blockIdx.z;
  A += bz * sA;
  B += bz * sB;

  // ---- fully-masked causal block: store -inf and exit (skips ~half of QK^T) ----
  if (causal && n0 > m0 + 255) {
    u16* Cb = (u16*)C + bz * sC;
    const uint32_t ninf2 = 0xFF80FF80u;
    uint4 qv; qv.x = ninf2; qv.y = ninf2; qv.z = ninf2; qv.w = ninf2;
#pragma unroll
    for (int i = 0; i < 16; ++i) {
      int v = tid + i * 256;           // 4096 uint4 = 256x128 bf16
      int row = v >> 4, col = (v & 15) * 8;
      *(uint4*)&Cb[(size_t)(m0 + row) * ldc + n0 + col] = qv;
    }
    return;
  }

  v8f acc[4][4] = {};

  // staging indices: A tile 1024 x b128 (4/thread), B tile 512 x b128 (2/thread)
  int arow[4], acol[4];
#pragma unroll
  for (int i = 0; i < 4; ++i) {
    int v = tid + i * 256;
    arow[i] = v >> 2; acol[i] = (v & 3) * 8;
  }
  int brow[2], bcol[2];
#pragma unroll
  for (int i = 0; i < 2; ++i) {
    int v = tid + i * 256;
    brow[i] = v >> 2; bcol[i] = (v & 3) * 8;
  }

  // incrementing global pointers
  const u16* aPtr[4];
#pragma unroll
  for (int i = 0; i < 4; ++i) aPtr[i] = A + (size_t)(m0 + arow[i]) * lda + acol[i];
  const u16* bPtr[2];
#pragma unroll
  for (int i = 0; i < 2; ++i) bPtr[i] = B + (size_t)(n0 + brow[i]) * ldb + bcol[i];

  const int nk = K >> 5;   // always >= 32 here

  auto issueStage = [&](int buf) {
#pragma unroll
    for (int i = 0; i < 4; ++i) {
      ASYNC_LD_B128(&As[buf][arow[i] * 40 + acol[i]], aPtr[i]);
      aPtr[i] += 32;
    }
#pragma unroll
    for (int i = 0; i < 2; ++i) {
      ASYNC_LD_B128(&Bs[buf][brow[i] * 40 + bcol[i]], bPtr[i]);
      bPtr[i] += 32;
    }
  };

  union Frag { v16bf v; uint4 q[2]; };
  auto compute = [&](int buf) {
    Frag b[4];
#pragma unroll
    for (int tn = 0; tn < 4; ++tn) {
      const u16* pbs = &Bs[buf][(wn * 64 + tn * 16 + l16) * 40];
      b[tn].q[0] = *(const uint4*)(pbs + hi * 16);
      b[tn].q[1] = *(const uint4*)(pbs + hi * 16 + 8);
    }
#pragma unroll
    for (int tm = 0; tm < 4; ++tm) {
      Frag a;
      const u16* pas = &As[buf][(wm * 64 + tm * 16 + l16) * 40];
      a.q[0] = *(const uint4*)(pas + hi * 16);
      a.q[1] = *(const uint4*)(pas + hi * 16 + 8);
#pragma unroll
      for (int tn = 0; tn < 4; ++tn)
        acc[tm][tn] = __builtin_amdgcn_wmma_f32_16x16x32_bf16(
            false, a.v, false, b[tn].v, (short)0, acc[tm][tn], false, false);
    }
  };

  // prologue: async-fill stages 0 and 1 (buffers 0, 1)
  issueStage(0);
  issueStage(1);

  int cur = 0, nn = 2;     // nn = buffer receiving stage ks+2
  for (int ks = 0; ks < nk - 1; ++ks) {
    WAIT_ASYNC(6);         // stage ks resident; stage ks+1 still in flight
    __syncthreads();
    if (ks + 2 < nk) issueStage(nn);  // buffer last read at iter ks-1
    compute(cur);
    cur = (cur == 2) ? 0 : cur + 1;
    nn  = (nn  == 2) ? 0 : nn  + 1;
  }
  // peeled final stage: drain all async ops
  WAIT_ASYNC(0);
  __syncthreads();
  compute(cur);

  // ---- epilogue: bias hoisted (4 loads), straight-line stores ----
  u16*   Cb = (u16*)C + bz * sC;
  float* Cf = (float*)C + bz * sC;
  float bv[4];
#pragma unroll
  for (int tn = 0; tn < 4; ++tn) bv[tn] = 0.f;
  if (bias) {
#pragma unroll
    for (int tn = 0; tn < 4; ++tn) bv[tn] = bias[n0 + wn * 64 + tn * 16 + l16];
  }
  const int rowBase = m0 + wm * 64 + hi * 8;
  const int colBase = n0 + wn * 64 + l16;
#pragma unroll
  for (int tm = 0; tm < 4; ++tm)
#pragma unroll
    for (int tn = 0; tn < 4; ++tn) {
      const int gcol = colBase + tn * 16;
#pragma unroll
      for (int r = 0; r < 8; ++r) {
        const int grow = rowBase + tm * 16 + r;
        float v = acc[tm][tn][r] * scale + bv[tn];
        if (causal && gcol > grow) v = -__builtin_inff();
        size_t offc = (size_t)grow * ldc + gcol;
        if (out_bf16) Cb[offc] = f2bf(v);
        else          Cf[offc] = v;
      }
    }
}

// ---------- V transpose: Vt[b][d][s] = qkv[b*S+s][2H + d] ----------
__global__ void k_transpose_v(const u16* __restrict__ qkv, u16* __restrict__ Vt,
                              int S, int H3, int H, int coff) {
  __shared__ u16 tile[32][33];
  int b  = blockIdx.z;
  int s0 = blockIdx.x * 32;
  int d0 = blockIdx.y * 32;
  const u16* src = qkv + (size_t)b * S * H3;
  u16*       dst = Vt  + (size_t)b * H * S;
  for (int i = threadIdx.y; i < 32; i += 8)
    tile[i][threadIdx.x] = src[(size_t)(s0 + i) * H3 + coff + d0 + threadIdx.x];
  __syncthreads();
  for (int i = threadIdx.y; i < 32; i += 8)
    dst[(size_t)(d0 + i) * S + s0 + threadIdx.x] = tile[threadIdx.x][i];
}

// ---------- in-place row softmax on bf16 scores (f32 math) ----------
__global__ __launch_bounds__(256) void k_softmax(u16* __restrict__ P, int S) {
  __shared__ float red[256];
  int row = blockIdx.x, b = blockIdx.y;
  u16* p = P + ((size_t)b * S + row) * S;
  float v[8];
  float m = -__builtin_inff();
#pragma unroll
  for (int i = 0; i < 8; ++i) {
    v[i] = bf2f(p[i * 256 + threadIdx.x]);
    m = fmaxf(m, v[i]);
  }
  red[threadIdx.x] = m;
  __syncthreads();
  for (int s = 128; s > 0; s >>= 1) {
    if (threadIdx.x < s) red[threadIdx.x] = fmaxf(red[threadIdx.x], red[threadIdx.x + s]);
    __syncthreads();
  }
  m = red[0];
  __syncthreads();
  float sum = 0.f;
#pragma unroll
  for (int i = 0; i < 8; ++i) { v[i] = __expf(v[i] - m); sum += v[i]; }
  red[threadIdx.x] = sum;
  __syncthreads();
  for (int s = 128; s > 0; s >>= 1) {
    if (threadIdx.x < s) red[threadIdx.x] += red[threadIdx.x + s];
    __syncthreads();
  }
  float inv = 1.0f / red[0];
#pragma unroll
  for (int i = 0; i < 8; ++i) p[i * 256 + threadIdx.x] = f2bf(v[i] * inv);
}

// ---------- launcher ----------
extern "C" void kernel_launch(void* const* d_in, const int* in_sizes, int n_in,
                              void* d_out, int out_size, void* d_ws, size_t ws_size,
                              hipStream_t stream) {
  const int B = 4, S = 2048, H = 1024;
  const int H3 = 3 * H;
  const int BS = B * S;  // 8192

  const float* x    = (const float*)d_in[0];
  const float* Wqkv = (const float*)d_in[1];
  const float* bqkv = (const float*)d_in[2];
  const float* Wout = (const float*)d_in[3];
  const float* bout = (const float*)d_in[4];
  float* out = (float*)d_out;

  char* ws = (char*)d_ws;
  size_t off = 0;
  auto carve = [&](size_t bytes) -> void* {
    void* p = ws + off;
    off += (bytes + 255) & ~(size_t)255;
    return p;
  };
  u16* xb    = (u16*)carve((size_t)BS * H  * 2);   // x in bf16
  u16* Wqkvb = (u16*)carve((size_t)H3 * H  * 2);   // Wqkv bf16
  u16* Woutb = (u16*)carve((size_t)H  * H  * 2);   // Wout bf16
  u16* qkv   = (u16*)carve((size_t)BS * H3 * 2);   // qkv bf16
  u16* Vt    = (u16*)carve((size_t)B * H * S * 2); // V^T bf16
  u16* Pr    = (u16*)carve((size_t)B * S * S * 2); // scores/probs bf16 (in-place softmax)
  u16* attn  = (u16*)carve((size_t)BS * H  * 2);   // attention output bf16
  (void)ws_size; (void)in_sizes; (void)n_in; (void)out_size;

  // 1) converts
  {
    long long n4;
    n4 = (long long)BS * H / 4;
    k_cvt<<<dim3((unsigned)((n4 + 255) / 256)), dim3(256), 0, stream>>>(x, xb, n4);
    n4 = (long long)H3 * H / 4;
    k_cvt<<<dim3((unsigned)((n4 + 255) / 256)), dim3(256), 0, stream>>>(Wqkv, Wqkvb, n4);
    n4 = (long long)H * H / 4;
    k_cvt<<<dim3((unsigned)((n4 + 255) / 256)), dim3(256), 0, stream>>>(Wout, Woutb, n4);
  }

  // 2) QKV projection: qkv[BS,3H] = xb @ Wqkv^T + bqkv   (bf16 out)
  k_gemm_nt_bf16<<<dim3(H3 / 128, BS / 256, 1), 256, 0, stream>>>(
      xb, Wqkvb, qkv, bqkv, BS, H3, H, H, H, H3, 0, 0, 0, 1.0f, 0, 1);

  // 3) V transpose
  k_transpose_v<<<dim3(S / 32, H / 32, B), dim3(32, 8), 0, stream>>>(
      qkv, Vt, S, H3, H, 2 * H);

  // 4) scores = scale * Q @ K^T with causal mask (per batch, bf16 out, L2-resident)
  const float scale = 1.0f / sqrtf((float)H);
  k_gemm_nt_bf16<<<dim3(S / 128, S / 256, B), 256, 0, stream>>>(
      qkv, qkv + H, Pr, nullptr, S, S, H, H3, H3, S,
      (long long)S * H3, (long long)S * H3, (long long)S * S, scale, 1, 1);

  // 5) row softmax in place
  k_softmax<<<dim3(S, B), 256, 0, stream>>>(Pr, S);

  // 6) attn_out = P @ V  (NT against Vt, per batch, bf16 out)
  k_gemm_nt_bf16<<<dim3(H / 128, S / 256, B), 256, 0, stream>>>(
      Pr, Vt, attn, nullptr, S, H, S, S, S, H,
      (long long)S * S, (long long)H * S, (long long)S * H, 1.0f, 0, 1);

  // 7) out = attn @ Wout^T + bout  (f32 out to d_out)
  k_gemm_nt_bf16<<<dim3(H / 128, BS / 256, 1), 256, 0, stream>>>(
      attn, Woutb, out, bout, BS, H, H, H, H, H, 0, 0, 0, 1.0f, 0, 0);
}
